// DecoderBlock_53077205844367
// MI455X (gfx1250) — matrix operand order repaired and checked by reading, not compiled, not run on hardware
//
#include <hip/hip_runtime.h>
#include <math.h>

typedef __attribute__((ext_vector_type(16))) _Float16 v16h;
typedef __attribute__((ext_vector_type(8)))  float    v8f;
typedef __attribute__((ext_vector_type(4)))  unsigned int u32x4;
typedef int v4i __attribute__((vector_size(16)));

#define D_MODEL 1024
#define N_HEADS 16
#define HEAD_D  64
#define BATCH   2
#define SEQ     2048
#define NTOK    (BATCH*SEQ)
#define DFF     (D_MODEL*4)
#define LN_EPS  1e-5f

// ---- CDNA5 async global->LDS DMA (ASYNCcnt), with safe fallback -------------
#if defined(__has_builtin)
#if __has_builtin(__builtin_amdgcn_global_load_async_to_lds_b128) && \
    __has_builtin(__builtin_amdgcn_s_wait_asynccnt)
#define ASYNC_LDS 1
#endif
#endif
#ifndef ASYNC_LDS
#define ASYNC_LDS 0
#endif

__device__ __forceinline__ void cp_g2l_16(_Float16* ldst, const _Float16* gsrc) {
#if ASYNC_LDS
  __builtin_amdgcn_global_load_async_to_lds_b128(
      (__attribute__((address_space(1))) v4i*)gsrc,
      (__attribute__((address_space(3))) v4i*)ldst, 0, 0);
#else
  *(u32x4*)ldst = *(const u32x4*)gsrc;
#endif
}
__device__ __forceinline__ void async_fence() {
#if ASYNC_LDS
  __builtin_amdgcn_s_wait_asynccnt(0);
#endif
}

union FragH { v16h v; _Float16 h[16]; u32x4 q[2]; };

__device__ __forceinline__ v8f wmma_f16(const FragH& a, const FragH& b, v8f c) {
  return __builtin_amdgcn_wmma_f32_16x16x32_f16(false, a.v, false, b.v,
                                                (short)0, c, false, false);
}

// ---------------------------------------------------------------- convert (plain)
__global__ void k_cvt_f16(const float* __restrict__ in, _Float16* __restrict__ out, int n) {
  int i = (blockIdx.x * blockDim.x + threadIdx.x) * 4;
  if (i + 3 < n) {
    float4 f = *(const float4*)(in + i);
    out[i+0] = (_Float16)f.x; out[i+1] = (_Float16)f.y;
    out[i+2] = (_Float16)f.z; out[i+3] = (_Float16)f.w;
  }
}

// ---------------------------------------------------------------- convert + transpose
// out[n*Krows + k] = (f16) in[k*Ncols + n]   (weights: [K,N] -> [N,K])
__global__ void k_cvt_t_f16(const float* __restrict__ in, _Float16* __restrict__ out,
                            int Krows, int Ncols) {
  __shared__ _Float16 tile[32][33];
  int kb = blockIdx.y * 32, nb = blockIdx.x * 32;
  #pragma unroll
  for (int i = 0; i < 4; ++i) {
    int k = kb + threadIdx.y + i * 8;
    tile[threadIdx.y + i * 8][threadIdx.x] =
        (_Float16)in[(size_t)k * Ncols + nb + threadIdx.x];
  }
  __syncthreads();
  #pragma unroll
  for (int i = 0; i < 4; ++i) {
    int n = nb + threadIdx.y + i * 8;
    out[(size_t)n * Krows + kb + threadIdx.x] = tile[threadIdx.x][threadIdx.y + i * 8];
  }
}

// ---------------------------------------------------------------- GEMM
// C[M,N] = A[M,K](f16) * Bt[N,K](f16, pre-transposed) + bias.
// 256 threads (8 waves), tile 128x128, BK=32, double-buffered LDS, async staging.
#define G_LDA 40   // padded LDS row stride (halves); 80B keeps 16B alignment
template<int RELU, bool WF, bool WH>
__launch_bounds__(256)
__global__ void k_gemm(const _Float16* __restrict__ A, const _Float16* __restrict__ Bt,
                       const float* __restrict__ bias,
                       float* __restrict__ outF, _Float16* __restrict__ outH,
                       int M, int N, int K) {
  __shared__ __align__(16) _Float16 As[2][128 * G_LDA];
  __shared__ __align__(16) _Float16 Bs[2][128 * G_LDA];
  const int tid = threadIdx.x, lane = tid & 31, w = tid >> 5;
  const int half = lane >> 4, l16 = lane & 15;
  const int wm = w >> 1, wn = w & 1;
  const int m0 = blockIdx.y * 128, n0 = blockIdx.x * 128;

  v8f z = {0.f,0.f,0.f,0.f,0.f,0.f,0.f,0.f};
  v8f c[2][4];
  #pragma unroll
  for (int mt = 0; mt < 2; ++mt)
    #pragma unroll
    for (int nt = 0; nt < 4; ++nt) c[mt][nt] = z;

  auto stage = [&](int buf, int k0) {
    int r = tid >> 2, c8 = (tid & 3) * 8;
    #pragma unroll
    for (int i = 0; i < 2; ++i) {
      int rr = r + i * 64;
      cp_g2l_16(&As[buf][rr * G_LDA + c8], &A[(size_t)(m0 + rr) * K + k0 + c8]);
      cp_g2l_16(&Bs[buf][rr * G_LDA + c8], &Bt[(size_t)(n0 + rr) * K + k0 + c8]);
    }
  };

  stage(0, 0);
  async_fence();
  __syncthreads();
  int buf = 0;
  for (int k0 = 0; k0 < K; k0 += 32) {
    if (k0 + 32 < K) stage(buf ^ 1, k0 + 32);
    FragH a[2], b[4];
    #pragma unroll
    for (int mt = 0; mt < 2; ++mt) {
      int mr = wm * 32 + mt * 16 + l16;
      a[mt].q[0] = *(const u32x4*)&As[buf][mr * G_LDA + half * 8];
      a[mt].q[1] = *(const u32x4*)&As[buf][mr * G_LDA + 16 + half * 8];
    }
    #pragma unroll
    for (int nt = 0; nt < 4; ++nt) {
      int nr = wn * 64 + nt * 16 + l16;
      b[nt].q[0] = *(const u32x4*)&Bs[buf][nr * G_LDA + half * 8];
      b[nt].q[1] = *(const u32x4*)&Bs[buf][nr * G_LDA + 16 + half * 8];
    }
    #pragma unroll
    for (int mt = 0; mt < 2; ++mt)
      #pragma unroll
      for (int nt = 0; nt < 4; ++nt)
        c[mt][nt] = wmma_f16(a[mt], b[nt], c[mt][nt]);
    async_fence();            // next-tile DMA done (after the math, full overlap)
    __syncthreads();
    buf ^= 1;
  }

  #pragma unroll
  for (int nt = 0; nt < 4; ++nt) {
    int n = n0 + wn * 64 + nt * 16 + l16;
    float bv = bias[n];
    #pragma unroll
    for (int mt = 0; mt < 2; ++mt) {
      #pragma unroll
      for (int v = 0; v < 8; ++v) {
        int m = m0 + wm * 32 + mt * 16 + v + half * 8;
        float val = c[mt][nt][v] + bv;
        if (RELU) val = fmaxf(val, 0.f);
        size_t idx = (size_t)m * N + n;
        if (WF) outF[idx] = val;
        if (WH) outH[idx] = (_Float16)val;
      }
    }
  }
}

// ---------------------------------------------------------------- V transpose
// Vt[bh][d(64)][s(2048)] = Vh[(b*SEQ+s)*D_MODEL + h*64 + d]
__global__ void k_transpose_v(const _Float16* __restrict__ Vh, _Float16* __restrict__ Vt) {
  int d  = threadIdx.x;
  int s  = blockIdx.x * 4 + threadIdx.y;
  int bh = blockIdx.y;
  int b = bh >> 4, h = bh & 15;
  _Float16 val = Vh[((size_t)(b * SEQ + s)) * D_MODEL + h * HEAD_D + d];
  Vt[((size_t)bh * HEAD_D + d) * SEQ + s] = val;
}

// ---------------------------------------------------------------- flash attention
// 4 waves/block, one 16-row q-tile per wave. KV chunk (128) staged in LDS,
// shared by all waves. Uniform chunk count per block (causal-safe numerically).
#define AT_CHUNK 128
#define KS_LD 72    // 64 + 8 pad (halves)
#define VS_LD 136   // 128 + 8 pad
#define PS_LD 136
__launch_bounds__(128)
__global__ void k_attn(const _Float16* __restrict__ Qh, const _Float16* __restrict__ Kh,
                       const _Float16* __restrict__ Vt, _Float16* __restrict__ Ctx,
                       int causal) {
  __shared__ __align__(16) _Float16 Ks[AT_CHUNK * KS_LD];   // 18KB
  __shared__ __align__(16) _Float16 Vs[HEAD_D * VS_LD];     // 17KB
  __shared__ __align__(16) _Float16 Ps[4][16 * PS_LD];      // 17KB
  const int tid = threadIdx.x, lane = tid & 31, w = tid >> 5;
  const int half = lane >> 4, l16 = lane & 15;
  const int bh = blockIdx.y, b = bh >> 4, h = bh & 15;
  const int q0 = blockIdx.x * 64 + w * 16;

  FragH aq[2];
  #pragma unroll
  for (int ks = 0; ks < 2; ++ks) {
    const _Float16* p = Qh + ((size_t)(b * SEQ + q0 + l16)) * D_MODEL
                          + h * HEAD_D + ks * 32 + half * 8;
    aq[ks].q[0] = *(const u32x4*)p;
    aq[ks].q[1] = *(const u32x4*)(p + 16);
  }

  v8f z = {0.f,0.f,0.f,0.f,0.f,0.f,0.f,0.f};
  v8f acc[4]; acc[0] = z; acc[1] = z; acc[2] = z; acc[3] = z;
  float rowM = -1e30f, rowL = 0.f;

  int nchunks = SEQ / AT_CHUNK;
  if (causal) nchunks = (blockIdx.x * 64 + 63) / AT_CHUNK + 1;

  for (int ci = 0; ci < nchunks; ++ci) {
    const int c0 = ci * AT_CHUNK;

    // cooperative stage: K chunk [128][64], V chunk [64][128] (from Vt)
    #pragma unroll
    for (int i = 0; i < 8; ++i) {
      int idx = tid + i * 128;
      int r = idx >> 3, c8 = (idx & 7) * 8;
      cp_g2l_16(&Ks[r * KS_LD + c8],
                &Kh[((size_t)(b * SEQ + c0 + r)) * D_MODEL + h * HEAD_D + c8]);
    }
    #pragma unroll
    for (int i = 0; i < 8; ++i) {
      int idx = tid + i * 128;
      int r = idx >> 4, c8 = (idx & 15) * 8;
      cp_g2l_16(&Vs[r * VS_LD + c8],
                &Vt[((size_t)bh * HEAD_D + r) * SEQ + c0 + c8]);
    }
    async_fence();
    __syncthreads();

    // scores S = Q K^T / 8, masked, raw f16 into Ps
    for (int n0 = 0; n0 < AT_CHUNK; n0 += 16) {
      v8f s = z;
      #pragma unroll
      for (int ks = 0; ks < 2; ++ks) {
        FragH bk;
        const _Float16* p = &Ks[(n0 + l16) * KS_LD + ks * 32 + half * 8];
        bk.q[0] = *(const u32x4*)p;
        bk.q[1] = *(const u32x4*)(p + 16);
        s = wmma_f16(aq[ks], bk, s);
      }
      int col = c0 + n0 + l16;
      #pragma unroll
      for (int v = 0; v < 8; ++v) {
        int m = v + half * 8;
        float val = s[v] * 0.125f;
        if (causal && col > q0 + m) val = -1e4f;
        Ps[w][m * PS_LD + n0 + l16] = (_Float16)val;
      }
    }
    __builtin_amdgcn_wave_barrier();

    // online softmax: lane pair (l16, l16+16) shares row l16, 64 cols each
    _Float16* sr = &Ps[w][l16 * PS_LD + half * 64];
    float cmax = -1e30f;
    for (int i = 0; i < 64; ++i) cmax = fmaxf(cmax, (float)sr[i]);
    cmax = fmaxf(cmax, __shfl_xor(cmax, 16, 32));
    float newM = fmaxf(rowM, cmax);
    float sc   = __expf(rowM - newM);
    float csum = 0.f;
    for (int i = 0; i < 64; ++i) {
      float e = __expf((float)sr[i] - newM);
      sr[i] = (_Float16)e;
      csum += e;
    }
    csum += __shfl_xor(csum, 16, 32);
    rowL = rowL * sc + csum;
    rowM = newM;
    __builtin_amdgcn_wave_barrier();

    // rescale accumulators by per-row factor
    #pragma unroll
    for (int v = 0; v < 8; ++v) {
      float f = __shfl(sc, v + half * 8, 32);
      #pragma unroll
      for (int nt = 0; nt < 4; ++nt) acc[nt][v] *= f;
    }

    // acc += P @ V (both operands in LDS)
    #pragma unroll
    for (int ks = 0; ks < AT_CHUNK / 32; ++ks) {
      FragH ap;
      const _Float16* pp = &Ps[w][l16 * PS_LD + ks * 32 + half * 8];
      ap.q[0] = *(const u32x4*)pp;
      ap.q[1] = *(const u32x4*)(pp + 16);
      #pragma unroll
      for (int nt = 0; nt < 4; ++nt) {
        FragH bv;
        const _Float16* vp = &Vs[(nt * 16 + l16) * VS_LD + ks * 32 + half * 8];
        bv.q[0] = *(const u32x4*)vp;
        bv.q[1] = *(const u32x4*)(vp + 16);
        acc[nt] = wmma_f16(ap, bv, acc[nt]);
      }
    }
    __syncthreads();
  }

  #pragma unroll
  for (int v = 0; v < 8; ++v) {
    float inv = 1.f / __shfl(rowL, v + half * 8, 32);
    int m = q0 + v + half * 8;
    #pragma unroll
    for (int nt = 0; nt < 4; ++nt) {
      Ctx[((size_t)(b * SEQ + m)) * D_MODEL + h * HEAD_D + nt * 16 + l16] =
          (_Float16)(acc[nt][v] * inv);
    }
  }
}

// ---------------------------------------------------------------- Add + LayerNorm
__launch_bounds__(256)
__global__ void k_add_ln(const float* __restrict__ a, const float* __restrict__ r,
                         const float* __restrict__ g, const float* __restrict__ beta,
                         float* __restrict__ outF, _Float16* __restrict__ outH) {
  __shared__ float red[256];
  __shared__ float row[D_MODEL];
  const int rr = blockIdx.x, tid = threadIdx.x;
  float s1 = 0.f, s2 = 0.f;
  for (int i = tid; i < D_MODEL; i += 256) {
    float v = a[(size_t)rr * D_MODEL + i] + r[(size_t)rr * D_MODEL + i];
    row[i] = v; s1 += v; s2 += v * v;
  }
  red[tid] = s1; __syncthreads();
  for (int o = 128; o > 0; o >>= 1) { if (tid < o) red[tid] += red[tid + o]; __syncthreads(); }
  float mean = red[0] * (1.f / D_MODEL);
  __syncthreads();
  red[tid] = s2; __syncthreads();
  for (int o = 128; o > 0; o >>= 1) { if (tid < o) red[tid] += red[tid + o]; __syncthreads(); }
  float var  = red[0] * (1.f / D_MODEL) - mean * mean;
  float rstd = rsqrtf(var + LN_EPS);
  for (int i = tid; i < D_MODEL; i += 256) {
    float v = (row[i] - mean) * rstd * g[i] + beta[i];
    outF[(size_t)rr * D_MODEL + i] = v;
    if (outH) outH[(size_t)rr * D_MODEL + i] = (_Float16)v;
  }
}

// ---------------------------------------------------------------- launch
extern "C" void kernel_launch(void* const* d_in, const int* in_sizes, int n_in,
                              void* d_out, int out_size, void* d_ws, size_t ws_size,
                              hipStream_t stream) {
  (void)in_sizes; (void)n_in; (void)out_size; (void)ws_size;
  const float* x    = (const float*)d_in[0];
  const float* enc  = (const float*)d_in[1];
  const float* wq_s = (const float*)d_in[4];  const float* bq_s = (const float*)d_in[5];
  const float* wk_s = (const float*)d_in[6];  const float* bk_s = (const float*)d_in[7];
  const float* wv_s = (const float*)d_in[8];  const float* bv_s = (const float*)d_in[9];
  const float* wo_s = (const float*)d_in[10]; const float* bo_s = (const float*)d_in[11];
  const float* wq_c = (const float*)d_in[12]; const float* bq_c = (const float*)d_in[13];
  const float* wk_c = (const float*)d_in[14]; const float* bk_c = (const float*)d_in[15];
  const float* wv_c = (const float*)d_in[16]; const float* bv_c = (const float*)d_in[17];
  const float* wo_c = (const float*)d_in[18]; const float* bo_c = (const float*)d_in[19];
  const float* ff1w = (const float*)d_in[20]; const float* ff1b = (const float*)d_in[21];
  const float* ff2w = (const float*)d_in[22]; const float* ff2b = (const float*)d_in[23];
  const float* ln1g = (const float*)d_in[24]; const float* ln1b = (const float*)d_in[25];
  const float* ln2g = (const float*)d_in[26]; const float* ln2b = (const float*)d_in[27];
  const float* ln3g = (const float*)d_in[28]; const float* ln3b = (const float*)d_in[29];

  char* ws = (char*)d_ws;
  auto alloc = [&](size_t bytes) -> char* {
    char* p = ws; ws += (bytes + 255) & ~(size_t)255; return p;
  };
  const size_t TD = (size_t)NTOK * D_MODEL;
  const size_t DD = (size_t)D_MODEL * D_MODEL;
  const size_t DF = (size_t)D_MODEL * DFF;

  _Float16* xh   = (_Float16*)alloc(TD * 2);
  _Float16* ench = (_Float16*)alloc(TD * 2);
  _Float16* wqsh = (_Float16*)alloc(DD * 2);
  _Float16* wksh = (_Float16*)alloc(DD * 2);
  _Float16* wvsh = (_Float16*)alloc(DD * 2);
  _Float16* wosh = (_Float16*)alloc(DD * 2);
  _Float16* wqch = (_Float16*)alloc(DD * 2);
  _Float16* wkch = (_Float16*)alloc(DD * 2);
  _Float16* wvch = (_Float16*)alloc(DD * 2);
  _Float16* woch = (_Float16*)alloc(DD * 2);
  _Float16* ff1h = (_Float16*)alloc(DF * 2);
  _Float16* ff2h = (_Float16*)alloc(DF * 2);
  _Float16* qh   = (_Float16*)alloc(TD * 2);
  _Float16* kh   = (_Float16*)alloc(TD * 2);
  _Float16* vh   = (_Float16*)alloc(TD * 2);
  _Float16* vth  = (_Float16*)alloc(TD * 2);
  _Float16* ctxh = (_Float16*)alloc(TD * 2);
  float*    tmpF = (float*)   alloc(TD * 4);
  float*    h1F  = (float*)   alloc(TD * 4);
  _Float16* h1H  = (_Float16*)alloc(TD * 2);
  float*    h2F  = (float*)   alloc(TD * 4);
  _Float16* h2H  = (_Float16*)alloc(TD * 2);
  _Float16* ffh  = (_Float16*)alloc((size_t)NTOK * DFF * 2);

  auto cvt = [&](const float* src, _Float16* dst, size_t n) {
    k_cvt_f16<<<(int)(n / 4 / 256), 256, 0, stream>>>(src, dst, (int)n);
  };
  auto cvtT = [&](const float* src, _Float16* dst, int Krows, int Ncols) {
    k_cvt_t_f16<<<dim3(Ncols / 32, Krows / 32), dim3(32, 8), 0, stream>>>(src, dst, Krows, Ncols);
  };
  auto gemmH = [&](const _Float16* A, const _Float16* B, const float* bias,
                   _Float16* oH, int M, int N, int K) {
    k_gemm<0, false, true><<<dim3(N / 128, M / 128), 256, 0, stream>>>(
        A, B, bias, nullptr, oH, M, N, K);
  };
  auto gemmHrelu = [&](const _Float16* A, const _Float16* B, const float* bias,
                       _Float16* oH, int M, int N, int K) {
    k_gemm<1, false, true><<<dim3(N / 128, M / 128), 256, 0, stream>>>(
        A, B, bias, nullptr, oH, M, N, K);
  };
  auto gemmF = [&](const _Float16* A, const _Float16* B, const float* bias,
                   float* oF, int M, int N, int K) {
    k_gemm<0, true, false><<<dim3(N / 128, M / 128), 256, 0, stream>>>(
        A, B, bias, oF, nullptr, M, N, K);
  };
  auto attn = [&](const _Float16* Q, const _Float16* K, _Float16* C, int causal) {
    k_transpose_v<<<dim3(SEQ / 4, BATCH * N_HEADS), dim3(64, 4), 0, stream>>>(vh, vth);
    k_attn<<<dim3(SEQ / 64, BATCH * N_HEADS), 128, 0, stream>>>(Q, K, vth, C, causal);
  };

  // convert activations; convert+transpose all weights
  cvt(x, xh, TD); cvt(enc, ench, TD);
  cvtT(wq_s, wqsh, D_MODEL, D_MODEL); cvtT(wk_s, wksh, D_MODEL, D_MODEL);
  cvtT(wv_s, wvsh, D_MODEL, D_MODEL); cvtT(wo_s, wosh, D_MODEL, D_MODEL);
  cvtT(wq_c, wqch, D_MODEL, D_MODEL); cvtT(wk_c, wkch, D_MODEL, D_MODEL);
  cvtT(wv_c, wvch, D_MODEL, D_MODEL); cvtT(wo_c, woch, D_MODEL, D_MODEL);
  cvtT(ff1w, ff1h, D_MODEL, DFF);     cvtT(ff2w, ff2h, DFF, D_MODEL);

  // ---- self attention
  gemmH(xh, wqsh, bq_s, qh, NTOK, D_MODEL, D_MODEL);
  gemmH(xh, wksh, bk_s, kh, NTOK, D_MODEL, D_MODEL);
  gemmH(xh, wvsh, bv_s, vh, NTOK, D_MODEL, D_MODEL);
  attn(qh, kh, ctxh, 1);
  gemmF(ctxh, wosh, bo_s, tmpF, NTOK, D_MODEL, D_MODEL);
  k_add_ln<<<NTOK, 256, 0, stream>>>(x, tmpF, ln1g, ln1b, h1F, h1H);

  // ---- cross attention
  gemmH(h1H, wqch, bq_c, qh, NTOK, D_MODEL, D_MODEL);
  gemmH(ench, wkch, bk_c, kh, NTOK, D_MODEL, D_MODEL);
  gemmH(ench, wvch, bv_c, vh, NTOK, D_MODEL, D_MODEL);
  attn(qh, kh, ctxh, 0);
  gemmF(ctxh, woch, bo_c, tmpF, NTOK, D_MODEL, D_MODEL);
  k_add_ln<<<NTOK, 256, 0, stream>>>(h1F, tmpF, ln2g, ln2b, h2F, h2H);

  // ---- FFN
  gemmHrelu(h2H, ff1h, ff1b, ffh, NTOK, DFF, D_MODEL);
  gemmF(ffh, ff2h, ff2b, tmpF, NTOK, D_MODEL, DFF);
  k_add_ln<<<NTOK, 256, 0, stream>>>(h2F, tmpF, ln3g, ln3b, (float*)d_out, nullptr);
}
